// GCNbase_9938554323112
// MI455X (gfx1250) — matrix operand order repaired and checked by reading, not compiled, not run on hardware
//
#include <hip/hip_runtime.h>
#include <hip/hip_bf16.h>
#include <stdint.h>

// ---------------------------------------------------------------------------
// 2-layer GCN for MI455X (gfx1250, wave32).
// GEMMs via v_wmma_f32_16x16x32_f16 (codegen-confirmed builtin), scatter via
// global_atomic_add_f32 (L2 atomic units), everything resident in 192MB L2.
// ---------------------------------------------------------------------------

typedef _Float16 h16v __attribute__((ext_vector_type(16)));
typedef _Float16 h8v  __attribute__((ext_vector_type(8)));
typedef _Float16 h4v  __attribute__((ext_vector_type(4)));
typedef float    f8v  __attribute__((ext_vector_type(8)));
typedef float    f4v  __attribute__((ext_vector_type(4)));

#define N_NODES 40000
#define N_EDGES 640000
#define F_IN    128
#define HID     128
#define C_OUT   40
#define C_PAD   48   // pad layer-2 output cols to 3 WMMA n-tiles

// ---------------- degree / normalization ----------------
__global__ void k_deg_init(float* __restrict__ deg) {
    int i = blockIdx.x * 256 + threadIdx.x;
    if (i < N_NODES) deg[i] = 1.0f;            // self-loop contribution
}

__global__ void k_deg_count(const long long* __restrict__ ei, float* __restrict__ deg) {
    int e = blockIdx.x * 256 + threadIdx.x;
    if (e < N_EDGES) atomicAdd(&deg[(int)ei[N_EDGES + e]], 1.0f);  // dst row
}

__global__ void k_deg_rsqrt(float* __restrict__ deg) {
    int i = blockIdx.x * 256 + threadIdx.x;
    if (i < N_NODES) deg[i] = rsqrtf(deg[i]);  // deg >= 1
}

// ---------------- fp32 -> fp16 conversions ----------------
__global__ void k_cvt_x(const float* __restrict__ x, _Float16* __restrict__ xh) {
    int t = blockIdx.x * 256 + threadIdx.x;    // one float4 per thread
    if (t >= N_NODES * F_IN / 4) return;
    f4v v = *(const f4v*)(x + (size_t)t * 4);
    h4v o;
    #pragma unroll
    for (int i = 0; i < 4; ++i) o[i] = (_Float16)v[i];
    *(h4v*)(xh + (size_t)t * 4) = o;
}

// W1 is [F_IN][HID] row-major; build W1t fp16 [HID][F_IN] (n-major, contiguous K)
__global__ void k_cvt_w1(const float* __restrict__ W1, _Float16* __restrict__ w1t) {
    int t = blockIdx.x * 256 + threadIdx.x;    // t = n*128 + k
    if (t >= HID * F_IN) return;
    int n = t >> 7, k = t & 127;
    w1t[t] = (_Float16)W1[(size_t)k * HID + n];
}

// W2 is [HID][C_OUT]; build W2t fp16 [C_PAD][HID], zero-padded columns
__global__ void k_cvt_w2(const float* __restrict__ W2, _Float16* __restrict__ w2t) {
    int t = blockIdx.x * 256 + threadIdx.x;    // t = n*128 + k
    if (t >= C_PAD * HID) return;
    int n = t >> 7, k = t & 127;
    w2t[t] = (n < C_OUT) ? (_Float16)W2[(size_t)k * C_OUT + n] : (_Float16)0.0f;
}

// ---------------- WMMA GEMM: C[M][Ntot] = A[M][128] * Bt[Ntot][128]^T --------
// One wave per 16x16 tile; blockDim.x = 32*(Ntot/16); gridDim.x = M/16.
// Fragment layouts per CDNA5 ISA 7.12.2 (16-bit A 16x32, B 32x16, f32 C/D).
__global__ void k_gemm_wmma(const _Float16* __restrict__ A,
                            const _Float16* __restrict__ Bt,
                            float* __restrict__ C,
                            int Ntot) {
    const int lane = threadIdx.x & 31;
    const int wave = threadIdx.x >> 5;         // n-tile index
    const int m0   = blockIdx.x * 16;
    const int n0   = wave * 16;
    const int hi   = lane >> 4;                // lane half-group
    const int li   = lane & 15;                // M (for A) / N (for B/C) index

    const _Float16* arow = A  + (size_t)(m0 + li) * 128;  // row m = li
    const _Float16* brow = Bt + (size_t)(n0 + li) * 128;  // col n = li

    f8v c = {};
    #pragma unroll
    for (int k0 = 0; k0 < 128; k0 += 32) {
        // A lane fragment: halfs 0-7 -> K = k0 + hi*8 .. +7 ; 8-15 -> +16
        h8v a0 = *(const h8v*)(arow + k0 + hi * 8);
        h8v a1 = *(const h8v*)(arow + k0 + hi * 8 + 16);
        // B lane fragment: halfs 0-7 -> K = k0 + hi*8 .. +7 ; 8-15 -> +16
        h8v b0 = *(const h8v*)(brow + k0 + hi * 8);
        h8v b1 = *(const h8v*)(brow + k0 + hi * 8 + 16);
        h16v a, b;
        #pragma unroll
        for (int i = 0; i < 8; ++i) {
            a[i] = a0[i]; a[i + 8] = a1[i];
            b[i] = b0[i]; b[i + 8] = b1[i];
        }
        c = __builtin_amdgcn_wmma_f32_16x16x32_f16(
                /*neg_a=*/false, a, /*neg_b=*/false, b,
                /*c_mod=*/(short)0, c, /*reuse_a=*/false, /*reuse_b=*/false);
    }
    // C/D layout: VGPR r -> M = r + hi*8, N = li
    float* crow = C + (size_t)m0 * Ntot + n0 + li;
    #pragma unroll
    for (int r = 0; r < 8; ++r)
        crow[(size_t)(r + hi * 8) * Ntot] = c[r];
}

// ---------------- aggregation: out = b + dinv^2 * xw (self loop) ------------
__global__ void k_agg_init(const float* __restrict__ xw, const float* __restrict__ dinv,
                           const float* __restrict__ bias, float* __restrict__ out,
                           int F, int Fs) {
    int t = blockIdx.x * 256 + threadIdx.x;    // one float4 per thread
    int tot = N_NODES * (F / 4);
    if (t >= tot) return;
    int i = t / (F / 4);
    int f = (t % (F / 4)) * 4;
    float d = dinv[i];
    float s = d * d;
    f4v v = *(const f4v*)(xw + (size_t)i * Fs + f);
    f4v b = *(const f4v*)(bias + f);
    *(f4v*)(out + (size_t)i * F + f) = v * s + b;
}

// out[dst] += xw[src] * dinv[src]*dinv[dst], chunk-fastest for coalescing
__global__ void k_agg_edges(const long long* __restrict__ ei, const float* __restrict__ xw,
                            const float* __restrict__ dinv, float* __restrict__ out,
                            int F, int Fs) {
    int t = blockIdx.x * 256 + threadIdx.x;
    int cpe = F / 4;                            // float4 chunks per edge
    int tot = N_EDGES * cpe;
    if (t >= tot) return;
    int e = t / cpe;
    int f = (t % cpe) * 4;
    int s = (int)ei[e];
    int d = (int)ei[N_EDGES + e];
    float norm = dinv[s] * dinv[d];
    f4v v = *(const f4v*)(xw + (size_t)s * Fs + f);
    float* o = out + (size_t)d * F + f;
    atomicAdd(o + 0, v[0] * norm);
    atomicAdd(o + 1, v[1] * norm);
    atomicAdd(o + 2, v[2] * norm);
    atomicAdd(o + 3, v[3] * norm);
}

// ---------------- relu + fp16 convert for layer-2 GEMM ----------------
__global__ void k_relu_cvt(const float* __restrict__ a, _Float16* __restrict__ hh) {
    int t = blockIdx.x * 256 + threadIdx.x;
    if (t >= N_NODES * HID / 4) return;
    f4v v = *(const f4v*)(a + (size_t)t * 4);
    h4v o;
    #pragma unroll
    for (int i = 0; i < 4; ++i) o[i] = (_Float16)fmaxf(v[i], 0.0f);
    *(h4v*)(hh + (size_t)t * 4) = o;
}

// ---------------- log_softmax over C_OUT=40 ----------------
__global__ void k_logsoftmax(const float* __restrict__ a, float* __restrict__ out) {
    int i = blockIdx.x * 256 + threadIdx.x;
    if (i >= N_NODES) return;
    const float* r = a + (size_t)i * C_OUT;
    float mx = r[0];
    #pragma unroll 4
    for (int f = 1; f < C_OUT; ++f) mx = fmaxf(mx, r[f]);
    float s = 0.0f;
    #pragma unroll 4
    for (int f = 0; f < C_OUT; ++f) s += __expf(r[f] - mx);
    float l = mx + __logf(s);
    float* o = out + (size_t)i * C_OUT;
    #pragma unroll 4
    for (int f = 0; f < C_OUT; ++f) o[f] = r[f] - l;
}

// ---------------------------------------------------------------------------
extern "C" void kernel_launch(void* const* d_in, const int* in_sizes, int n_in,
                              void* d_out, int out_size, void* d_ws, size_t ws_size,
                              hipStream_t stream) {
    const float*      x  = (const float*)d_in[0];
    const long long*  ei = (const long long*)d_in[1];
    const float*      W1 = (const float*)d_in[2];
    const float*      b1 = (const float*)d_in[3];
    const float*      W2 = (const float*)d_in[4];
    const float*      b2 = (const float*)d_in[5];
    float*            out = (float*)d_out;

    // carve workspace (all 256B aligned)
    char* p = (char*)d_ws;
    auto carve = [&](size_t bytes) -> void* {
        void* r = (void*)p;
        p += (bytes + 255) & ~(size_t)255;
        return r;
    };
    float*     deg  = (float*)    carve((size_t)N_NODES * 4);            // -> dinv
    _Float16*  xh   = (_Float16*) carve((size_t)N_NODES * F_IN * 2);
    _Float16*  w1t  = (_Float16*) carve((size_t)HID * F_IN * 2);
    _Float16*  w2t  = (_Float16*) carve((size_t)C_PAD * HID * 2);
    float*     xw1  = (float*)    carve((size_t)N_NODES * HID * 4);
    float*     agg1 = (float*)    carve((size_t)N_NODES * HID * 4);
    _Float16*  hh   = (_Float16*) carve((size_t)N_NODES * HID * 2);
    float*     xw2  = (float*)    carve((size_t)N_NODES * C_PAD * 4);
    float*     agg2 = (float*)    carve((size_t)N_NODES * C_OUT * 4);

    // degree + normalization
    k_deg_init <<<(N_NODES + 255) / 256, 256, 0, stream>>>(deg);
    k_deg_count<<<(N_EDGES + 255) / 256, 256, 0, stream>>>(ei, deg);
    k_deg_rsqrt<<<(N_NODES + 255) / 256, 256, 0, stream>>>(deg);

    // fp16 operand prep
    k_cvt_x <<<(N_NODES * F_IN / 4 + 255) / 256, 256, 0, stream>>>(x, xh);
    k_cvt_w1<<<(HID * F_IN + 255) / 256, 256, 0, stream>>>(W1, w1t);
    k_cvt_w2<<<(C_PAD * HID + 255) / 256, 256, 0, stream>>>(W2, w2t);

    // layer 1: GEMM (2500 m-tiles x 8 n-tiles, 8 waves/block) + aggregate + relu
    k_gemm_wmma<<<N_NODES / 16, 256, 0, stream>>>(xh, w1t, xw1, HID);
    k_agg_init <<<(N_NODES * (HID / 4) + 255) / 256, 256, 0, stream>>>(xw1, deg, b1, agg1, HID, HID);
    k_agg_edges<<<(N_EDGES * (HID / 4) + 255) / 256, 256, 0, stream>>>(ei, xw1, deg, agg1, HID, HID);
    k_relu_cvt <<<(N_NODES * HID / 4 + 255) / 256, 256, 0, stream>>>(agg1, hh);

    // layer 2: GEMM (3 n-tiles -> 96 threads/block) + aggregate
    k_gemm_wmma<<<N_NODES / 16, 96, 0, stream>>>(hh, w2t, xw2, C_PAD);
    k_agg_init <<<(N_NODES * (C_OUT / 4) + 255) / 256, 256, 0, stream>>>(xw2, deg, b2, agg2, C_OUT, C_PAD);
    k_agg_edges<<<(N_EDGES * (C_OUT / 4) + 255) / 256, 256, 0, stream>>>(ei, xw2, deg, agg2, C_OUT, C_PAD);

    // log_softmax -> d_out
    k_logsoftmax<<<(N_NODES + 255) / 256, 256, 0, stream>>>(agg2, out);
}
